// ASFormerLayer_73512660238630
// MI455X (gfx1250) — matrix operand order repaired and checked by reading, not compiled, not run on hardware
//
#include <hip/hip_runtime.h>
#include <hip/hip_bf16.h>
#include <math.h>

// ---------------------------------------------------------------------------
// ASFormer layer for MI455X (gfx1250): all GEMM-shaped math through
// v_wmma_f32_16x16x32_f16 (wave32, 16x16x32 tiles, f32 accumulate).
// GEMM uses a 32x64 macro-tile per wave: 8 WMMAs per k-step over
// 6 tile loads (2 A + 4 B) for ~2.7x the compute-per-byte of a naive tile.
// ---------------------------------------------------------------------------

typedef _Float16 half_t;
typedef __attribute__((ext_vector_type(16))) _Float16 v16h;
typedef __attribute__((ext_vector_type(8)))  float    v8f;

#define BB 2
#define SS 2048
#define DD 512
#define HH 8
#define HD 64
#define FF 2048
#define QKV_LD 1536
#define WIN 64
#define NEGINF (-1e30f)

static __device__ __forceinline__ v8f wmma_f16f32(v16h a, v16h b, v8f c) {
    // (neg_a, A, neg_b, B, c_mod, C, reuse_a, reuse_b)
    return __builtin_amdgcn_wmma_f32_16x16x32_f16(false, a, false, b, (short)0, c,
                                                  false, false);
}

// A-matrix (16x32 f16, MxK) loader, row-major source with leading dim ld.
// Layout: lanes 0-15 -> M=lane, K = 0..15;
//         lanes 16-31 -> M=lane-16, K = 16..31.
static __device__ __forceinline__ v16h load_a_g(const half_t* base, int ld,
                                                int m0, int k0) {
    int l  = threadIdx.x;
    int m  = m0 + (l & 15);
    int kb = k0 + ((l >> 4) << 4);
    const half_t* p = base + (size_t)m * ld + kb;
    v16h a;
#pragma unroll
    for (int j = 0; j < 8; ++j) { a[j] = p[j]; a[j + 8] = p[j + 8]; }
    return a;
}

// B-matrix (32x16 f16, KxN) loader from a row-major [N,K] weight (i.e. W
// as stored by torch, y = x @ W^T).  Lane l holds column n = n0 + (l&15),
// contiguous K values starting at k0 + (l>>4)*16.
static __device__ __forceinline__ v16h load_b_w(const half_t* w, int K,
                                                int n0, int k0) {
    int l  = threadIdx.x;
    int n  = n0 + (l & 15);
    int kb = k0 + ((l >> 4) << 4);
    const half_t* p = w + (size_t)n * K + kb;
    v16h b;
#pragma unroll
    for (int j = 0; j < 16; ++j) b[j] = p[j];
    return b;
}

// ---------------------------------------------------------------------------
// f32 -> f16 convert (weights)
// ---------------------------------------------------------------------------
__global__ void cvt_f32_f16(const float* __restrict__ in,
                            half_t* __restrict__ out, int n) {
    int i = blockIdx.x * blockDim.x + threadIdx.x;
    if (i < n) out[i] = (half_t)in[i];
}

// ---------------------------------------------------------------------------
// LayerNorm, wave per row (D=512 -> 16 f32 per lane).  Optional second LN
// (inner norm of the conv FFN).  Writes f32 and/or f16.
// ---------------------------------------------------------------------------
__global__ __launch_bounds__(32) void ln_kernel(
    const float* __restrict__ x, const float* __restrict__ g1,
    const float* __restrict__ b1, const float* __restrict__ g2,
    const float* __restrict__ b2, float* __restrict__ outf,
    half_t* __restrict__ outh) {
    const int row  = blockIdx.x;
    const int lane = threadIdx.x;
    float v[16];
#pragma unroll
    for (int i = 0; i < 16; ++i) v[i] = x[(size_t)row * DD + i * 32 + lane];

#pragma unroll 1
    for (int pass = 0; pass < 2; ++pass) {
        const float* g = (pass == 0) ? g1 : g2;
        const float* b = (pass == 0) ? b1 : b2;
        if (g == nullptr) break;
        float s = 0.f;
#pragma unroll
        for (int i = 0; i < 16; ++i) s += v[i];
#pragma unroll
        for (int o = 16; o >= 1; o >>= 1) s += __shfl_xor(s, o, 32);
        float mu = s * (1.0f / DD);
        float t = 0.f;
#pragma unroll
        for (int i = 0; i < 16; ++i) { float d = v[i] - mu; t += d * d; }
#pragma unroll
        for (int o = 16; o >= 1; o >>= 1) t += __shfl_xor(t, o, 32);
        float rs = rsqrtf(t * (1.0f / DD) + 1e-5f);
#pragma unroll
        for (int i = 0; i < 16; ++i) {
            int c = i * 32 + lane;
            v[i] = (v[i] - mu) * rs * g[c] + b[c];
        }
    }
#pragma unroll
    for (int i = 0; i < 16; ++i) {
        int c = i * 32 + lane;
        if (outf) outf[(size_t)row * DD + c] = v[i];
        if (outh) outh[(size_t)row * DD + c] = (half_t)v[i];
    }
}

// ---------------------------------------------------------------------------
// WMMA GEMM, 32x64 macro-tile per wave (2 M-subtiles x 4 N-subtiles).
// C[M,N] = A[M,K](f16) * W[N,K]^T(f16) + bias, fused epilogue:
// optional residual add (f32 [M,N]), exact GELU, per-row mask multiply,
// f16 and/or f32 stores.  Requires M%32==0, N%64==0, K%32==0.
// ---------------------------------------------------------------------------
__global__ __launch_bounds__(32) void gemm_wmma(
    const half_t* __restrict__ A, const half_t* __restrict__ W,
    const float* __restrict__ bias, const float* __restrict__ residual,
    const unsigned char* __restrict__ rowmask, float* __restrict__ outf,
    half_t* __restrict__ outh, int M, int N, int K, int do_gelu) {
    const int m0 = blockIdx.x * 32;
    const int n0 = blockIdx.y * 64;
    v8f acc[2][4];
#pragma unroll
    for (int mt = 0; mt < 2; ++mt)
#pragma unroll
        for (int nt = 0; nt < 4; ++nt) acc[mt][nt] = {};

#pragma unroll 1
    for (int ks = 0; ks < K; ks += 32) {
        v16h a0 = load_a_g(A, K, m0, ks);
        v16h a1 = load_a_g(A, K, m0 + 16, ks);
        // prefetch next k-step (lowers to global_prefetch_b8; no counters)
        if (ks + 32 < K) {
            int l = threadIdx.x;
            __builtin_prefetch(A + (size_t)(m0 + (l & 15)) * K + ks + 32, 0, 1);
            __builtin_prefetch(W + (size_t)(n0 + (l & 15)) * K + ks + 32, 0, 1);
        }
#pragma unroll
        for (int nt = 0; nt < 4; ++nt) {
            v16h b = load_b_w(W, K, n0 + nt * 16, ks);
            acc[0][nt] = wmma_f16f32(a0, b, acc[0][nt]);
            acc[1][nt] = wmma_f16f32(a1, b, acc[1][nt]);
        }
    }

    const int l  = threadIdx.x;
    const int hi = l >> 4;
#pragma unroll
    for (int mt = 0; mt < 2; ++mt) {
#pragma unroll
        for (int nt = 0; nt < 4; ++nt) {
            const int n  = n0 + nt * 16 + (l & 15);
            const float bv = bias ? bias[n] : 0.f;
#pragma unroll
            for (int r = 0; r < 8; ++r) {
                int m = m0 + mt * 16 + r + 8 * hi;
                float v = acc[mt][nt][r] + bv;
                if (residual) v += residual[(size_t)m * N + n];
                if (do_gelu)  v = 0.5f * v * (1.0f + erff(v * 0.70710678118654752f));
                if (rowmask)  v *= (float)rowmask[m];
                if (outh) outh[(size_t)m * N + n] = (half_t)v;
                if (outf) outf[(size_t)m * N + n] = v;
            }
        }
    }
}

// ---------------------------------------------------------------------------
// Banded local attention.  One wave per (b, h, 16-query tile).
// Window +-64 -> 144 keys, padded to 160 (10 tiles of 16).
//   scores = (Q K^T) * 1/sqrt(64), band + key-padding mask, softmax,
//   ctx = P V.  All matmuls via WMMA; P transposed through LDS into the
//   A-register layout; V staged in LDS.
// ---------------------------------------------------------------------------
__global__ __launch_bounds__(32) void attn_kernel(
    const half_t* __restrict__ qkv, const unsigned char* __restrict__ mask,
    half_t* __restrict__ ctx) {
    __shared__ float  sc[16][160];
    __shared__ half_t pbuf[16][160];
    __shared__ half_t vtile[160][64];
    __shared__ float  rsum[16];

    const int lane = threadIdx.x;
    const int bh   = blockIdx.y;
    const int b    = bh / HH;
    const int h    = bh % HH;
    const int q0   = blockIdx.x * 16;
    const int kbase = q0 - WIN;   // 16-aligned

    // init scores to -inf (tiles outside [0,S) stay masked)
    for (int i = lane; i < 16 * 160; i += 32)
        (&sc[0][0])[i] = NEGINF;

    // stage V window into LDS (zero-fill out-of-range keys)
    {
        unsigned int* vw = reinterpret_cast<unsigned int*>(&vtile[0][0]);
        for (int i = lane; i < 160 * 32; i += 32) {   // 32 dwords per key row
            int k  = i >> 5;
            int dp = i & 31;
            int key = kbase + k;
            unsigned int w = 0u;
            if (key >= 0 && key < SS) {
                const half_t* p = qkv + (size_t)(b * SS + key) * QKV_LD +
                                  1024 + h * HD + dp * 2;
                w = *reinterpret_cast<const unsigned int*>(p);
            }
            vw[i] = w;
        }
    }

    // Q tile in A layout (two 16x32 k-steps over head_dim 64)
    const half_t* qbase = qkv + (size_t)(b * SS + q0) * QKV_LD + h * HD;
    v16h qa0 = load_a_g(qbase, QKV_LD, 0, 0);
    v16h qa1 = load_a_g(qbase, QKV_LD, 0, 32);

    __syncthreads();

    // ---- scores ----
#pragma unroll 1
    for (int t = 0; t < 9; ++t) {
        int k0 = kbase + t * 16;
        if (k0 < 0 || k0 + 16 > SS) continue;   // uniform across the wave
        v8f c = {};
        {
            // B = K^T : lane holds key column n = l&15, contiguous dims
            const int n  = lane & 15;
            const half_t* krow = qkv + (size_t)(b * SS + k0 + n) * QKV_LD +
                                 512 + h * HD;
            v16h bm;
            int kd = (lane >> 4) << 4;
#pragma unroll
            for (int j = 0; j < 16; ++j) bm[j] = krow[kd + j];
            c = wmma_f16f32(qa0, bm, c);
#pragma unroll
            for (int j = 0; j < 16; ++j) bm[j] = krow[32 + kd + j];
            c = wmma_f16f32(qa1, bm, c);
        }
        const int n   = lane & 15;
        const int hi  = lane >> 4;
        const int key = k0 + n;
        const int col = t * 16 + n;
        const bool kp = mask[(size_t)b * SS + key] != 0;
#pragma unroll
        for (int r = 0; r < 8; ++r) {
            int q = q0 + r + 8 * hi;
            int dist = q > key ? q - key : key - q;
            bool ok = (dist <= WIN) && kp;
            sc[r + 8 * hi][col] = ok ? c[r] * 0.125f : NEGINF;
        }
    }
    __syncthreads();

    // ---- softmax (2 lanes per row, 80 cols each) ----
    {
        const int row = lane & 15;
        const int hsel = lane >> 4;
        float mx = NEGINF;
#pragma unroll 1
        for (int j = 0; j < 80; ++j)
            mx = fmaxf(mx, sc[row][hsel * 80 + j]);
        mx = fmaxf(mx, __shfl_xor(mx, 16, 32));
        float sum = 0.f;
#pragma unroll 1
        for (int j = 0; j < 80; ++j) {
            int c = hsel * 80 + j;
            float e = __expf(sc[row][c] - mx);
            pbuf[row][c] = (half_t)e;
            sum += e;
        }
        sum += __shfl_xor(sum, 16, 32);
        if (lane < 16) rsum[row] = fmaxf(sum, 1e-20f);
    }
    __syncthreads();

    // ---- ctx = P V ----
#pragma unroll 1
    for (int nt = 0; nt < 4; ++nt) {
        v8f acc = {};
#pragma unroll 1
        for (int ks = 0; ks < 5; ++ks) {
            // A = P from LDS in the A register layout
            v16h a;
            {
                int m  = lane & 15;
                int kb = ks * 32 + ((lane >> 4) << 4);
#pragma unroll
                for (int j = 0; j < 8; ++j) {
                    a[j]     = pbuf[m][kb + j];
                    a[j + 8] = pbuf[m][kb + 8 + j];
                }
            }
            // B = V from LDS, lane holds dim column d, strided over keys
            v16h bm;
            {
                int d  = nt * 16 + (lane & 15);
                int kk = ks * 32 + ((lane >> 4) << 4);
#pragma unroll
                for (int j = 0; j < 16; ++j) bm[j] = vtile[kk + j][d];
            }
            acc = wmma_f16f32(a, bm, acc);
        }
        const int d  = nt * 16 + (lane & 15);
        const int hi = lane >> 4;
#pragma unroll
        for (int r = 0; r < 8; ++r) {
            int m = r + 8 * hi;
            float v = acc[r] / rsum[m];
            ctx[(size_t)(b * SS + q0 + m) * DD + h * HD + d] = (half_t)v;
        }
    }
}

// ---------------------------------------------------------------------------
// Depthwise conv k=3 along S (zero pad), f16 out.
// ---------------------------------------------------------------------------
__global__ void dwconv_kernel(const float* __restrict__ y,
                              const float* __restrict__ w,
                              const float* __restrict__ bias,
                              half_t* __restrict__ out) {
    int idx = blockIdx.x * blockDim.x + threadIdx.x;
    if (idx >= BB * SS * DD) return;
    int c   = idx % DD;
    int row = idx / DD;
    int s   = row % SS;
    float acc = bias[c];
    if (s > 0)      acc += y[(size_t)(row - 1) * DD + c] * w[c * 3 + 0];
    acc += y[(size_t)row * DD + c] * w[c * 3 + 1];
    if (s + 1 < SS) acc += y[(size_t)(row + 1) * DD + c] * w[c * 3 + 2];
    out[idx] = (half_t)acc;
}

// ---------------------------------------------------------------------------
// Host-side orchestration
// ---------------------------------------------------------------------------
static inline size_t align256(size_t x) { return (x + 255) & ~(size_t)255; }

extern "C" void kernel_launch(void* const* d_in, const int* in_sizes, int n_in,
                              void* d_out, int out_size, void* d_ws, size_t ws_size,
                              hipStream_t stream) {
    const float* x      = (const float*)d_in[0];
    const unsigned char* mask = (const unsigned char*)d_in[1];  // jnp.bool_
    const float* ln_a_g = (const float*)d_in[2];
    const float* ln_a_b = (const float*)d_in[3];
    const float* w_qkv  = (const float*)d_in[4];
    const float* b_qkv  = (const float*)d_in[5];
    const float* w_o    = (const float*)d_in[6];
    const float* b_o    = (const float*)d_in[7];
    const float* ln_f_g = (const float*)d_in[8];
    const float* ln_f_b = (const float*)d_in[9];
    const float* ln_c_g = (const float*)d_in[10];
    const float* ln_c_b = (const float*)d_in[11];
    const float* dw_w   = (const float*)d_in[12];
    const float* dw_b   = (const float*)d_in[13];
    const float* pwi_w  = (const float*)d_in[14];
    const float* pwi_b  = (const float*)d_in[15];
    const float* pwo_w  = (const float*)d_in[16];
    const float* pwo_b  = (const float*)d_in[17];
    float* out = (float*)d_out;

    const int M = BB * SS;   // 4096 rows

    // workspace carve-up
    char* p = (char*)d_ws;
    size_t off = 0;
    auto take = [&](size_t bytes) { char* q = p + off; off = align256(off + bytes); return q; };
    half_t* a_ln   = (half_t*)take((size_t)M * DD * 2);
    half_t* qkv_h  = (half_t*)take((size_t)M * QKV_LD * 2);
    half_t* ctx_h  = (half_t*)take((size_t)M * DD * 2);
    float*  x1     = (float*) take((size_t)M * DD * 4);
    float*  yln    = (float*) take((size_t)M * DD * 4);
    half_t* conv_h = (half_t*)take((size_t)M * DD * 2);
    half_t* ffh    = (half_t*)take((size_t)M * FF * 2);
    half_t* wqkv_h = (half_t*)take((size_t)QKV_LD * DD * 2);
    half_t* wo_h   = (half_t*)take((size_t)DD * DD * 2);
    half_t* pwi_h  = (half_t*)take((size_t)FF * DD * 2);
    half_t* pwo_h  = (half_t*)take((size_t)DD * FF * 2);
    (void)ws_size;

    // 1) weights -> f16
    {
        int n;
        n = QKV_LD * DD; cvt_f32_f16<<<(n + 255) / 256, 256, 0, stream>>>(w_qkv, wqkv_h, n);
        n = DD * DD;     cvt_f32_f16<<<(n + 255) / 256, 256, 0, stream>>>(w_o,   wo_h,   n);
        n = FF * DD;     cvt_f32_f16<<<(n + 255) / 256, 256, 0, stream>>>(pwi_w, pwi_h,  n);
        n = DD * FF;     cvt_f32_f16<<<(n + 255) / 256, 256, 0, stream>>>(pwo_w, pwo_h,  n);
    }

    // 2) attention pre-norm -> f16
    ln_kernel<<<M, 32, 0, stream>>>(x, ln_a_g, ln_a_b, nullptr, nullptr,
                                    nullptr, a_ln);

    // 3) QKV projection -> f16 [M, 1536]
    gemm_wmma<<<dim3(M / 32, QKV_LD / 64), 32, 0, stream>>>(
        a_ln, wqkv_h, b_qkv, nullptr, nullptr, nullptr, qkv_h,
        M, QKV_LD, DD, 0);

    // 4) banded attention -> ctx f16 [M, 512]
    attn_kernel<<<dim3(SS / 16, BB * HH), 32, 0, stream>>>(qkv_h, mask, ctx_h);

    // 5) out projection + residual -> x1 f32
    gemm_wmma<<<dim3(M / 32, DD / 64), 32, 0, stream>>>(
        ctx_h, wo_h, b_o, x, nullptr, x1, nullptr, M, DD, DD, 0);

    // 6) outer + inner LN -> yln f32
    ln_kernel<<<M, 32, 0, stream>>>(x1, ln_f_g, ln_f_b, ln_c_g, ln_c_b,
                                    yln, nullptr);

    // 7) depthwise conv3 -> f16
    {
        int n = M * DD;
        dwconv_kernel<<<(n + 255) / 256, 256, 0, stream>>>(yln, dw_w, dw_b, conv_h);
    }

    // 8) pointwise-in + GELU -> ffh f16 [M, 2048]
    gemm_wmma<<<dim3(M / 32, FF / 64), 32, 0, stream>>>(
        conv_h, pwi_h, pwi_b, nullptr, nullptr, nullptr, ffh,
        M, FF, DD, 1);

    // 9) pointwise-out + residual + row mask -> d_out f32
    gemm_wmma<<<dim3(M / 32, DD / 64), 32, 0, stream>>>(
        ffh, pwo_h, pwo_b, x1, mask, out, nullptr, M, DD, FF, 0);

    (void)in_sizes; (void)n_in; (void)out_size;
}
